// GraphClassificationModel_51453708206168
// MI455X (gfx1250) — matrix operand order repaired and checked by reading, not compiled, not run on hardware
//
#include <hip/hip_runtime.h>
#include <math.h>
#include <stdint.h>

typedef __attribute__((ext_vector_type(16))) __bf16        v16bf;
typedef __attribute__((ext_vector_type(8)))  float         v8f;
typedef __attribute__((ext_vector_type(4)))  unsigned int  u32x4;

#define NEG_SLOPE 0.2f

__device__ __forceinline__ unsigned short f2bf(float x) {
  unsigned int u = __float_as_uint(x);
  unsigned int r = (u + 0x7FFFu + ((u >> 16) & 1u)) >> 16;   // RNE
  return (unsigned short)r;
}

__device__ __forceinline__ void atomicMaxF(float* addr, float val) {
  if (val >= 0.0f) atomicMax((int*)addr, __float_as_int(val));
  else             atomicMin((unsigned int*)addr, __float_as_uint(val));
}

// ---------------- WMMA GEMM: C[M16 x Nc] = A(bf16 M16xK) * Bt(bf16 Nc x K)^T (+bias) ---------
// A row-major ld=K ; Bt holds B transposed (column n contiguous over K). One 16x16 tile / wave.
__global__ void gemm_bf16_wmma(const unsigned short* __restrict__ A,
                               const unsigned short* __restrict__ Bt,
                               float* __restrict__ C,
                               int M16, int K, int Nc,
                               const float* __restrict__ bias) {
  const int tilesN = Nc >> 4;
  const int tilesM = M16 >> 4;
  int tile = blockIdx.x * 8 + (threadIdx.x >> 5);
  if (tile >= tilesM * tilesN) return;            // wave-uniform exit (EXEC all-1 for WMMA)
  int mt = tile / tilesN, nt = tile - mt * tilesN;
  int lane = threadIdx.x & 31;
  int lo = lane & 15, hi = lane >> 4;

  // ISA 16-bit A 16x32 layout: lanes0-15 K=0..7,16..23 ; lanes16-31 K=8..15,24..31
  const unsigned short* ap = A  + (size_t)(mt * 16 + lo) * K + hi * 8;
  // ISA 16-bit B 32x16 layout: lane holds column n, lanes0-15 K=0..7,16..23 ; lanes16-31 K=8..15,24..31
  const unsigned short* bp = Bt + (size_t)(nt * 16 + lo) * K + hi * 8;

  v8f acc = {};
  for (int k0 = 0; k0 < K; k0 += 32) {
    union { u32x4 q[2]; v16bf v; } a, b;
    a.q[0] = *(const u32x4*)(ap + k0);        // K = kbase+0..7
    a.q[1] = *(const u32x4*)(ap + k0 + 16);   // K = kbase+16..23
    b.q[0] = *(const u32x4*)(bp + k0);        // K = kbase+0..7
    b.q[1] = *(const u32x4*)(bp + k0 + 16);   // K = kbase+16..23
    acc = __builtin_amdgcn_wmma_f32_16x16x32_bf16(false, a.v, false, b.v,
                                                  (short)0, acc, false, false);
  }
  int col = nt * 16 + lo;
  float bs = bias ? bias[col] : 0.0f;
#pragma unroll
  for (int v = 0; v < 8; ++v) {
    int row = mt * 16 + hi * 8 + v;
    C[(size_t)row * Nc + col] = acc[v] + bs;
  }
}

// ---------------- elementwise / setup kernels ----------------
__global__ void x_to_bf(const float* __restrict__ x, unsigned short* __restrict__ hbf,
                        int N, int N16) {
  int idx = blockIdx.x * blockDim.x + threadIdx.x;
  if (idx >= N16 * 32) return;
  int node = idx >> 5, c = idx & 31;
  float v = (node < N && c < 3) ? x[node * 3 + c] : 0.0f;
  hbf[idx] = f2bf(v);
}

__global__ void wt_to_bf(const float* __restrict__ W, unsigned short* __restrict__ Wt,
                         int Cin, int Kpad, int Cout) {
  int idx = blockIdx.x * blockDim.x + threadIdx.x;
  if (idx >= Cout * Kpad) return;
  int o = idx / Kpad, k = idx - o * Kpad;
  float v = (k < Cin) ? W[(size_t)k * Cout + o] : 0.0f;
  Wt[idx] = f2bf(v);
}

__global__ void f_to_bf(const float* __restrict__ src, unsigned short* __restrict__ dst, int n) {
  int i = blockIdx.x * blockDim.x + threadIdx.x;
  if (i < n) dst[i] = f2bf(src[i]);
}

__global__ void deg_loop(const int* __restrict__ dstA, const float* __restrict__ ea,
                         float* __restrict__ deg, float* __restrict__ loops, int E) {
  int e = blockIdx.x * blockDim.x + threadIdx.x;
  if (e >= E) return;
  int d = dstA[e];
  atomicAdd(&deg[d], 1.0f);
  atomicAdd(&loops[d * 3 + 0], ea[e * 3 + 0]);
  atomicAdd(&loops[d * 3 + 1], ea[e * 3 + 1]);
  atomicAdd(&loops[d * 3 + 2], ea[e * 3 + 2]);
}

__global__ void loop_div(float* __restrict__ loops, const float* __restrict__ deg, int N) {
  int v = blockIdx.x * blockDim.x + threadIdx.x;
  if (v >= N) return;
  float d = fmaxf(deg[v], 1.0f);
  loops[v * 3 + 0] /= d; loops[v * 3 + 1] /= d; loops[v * 3 + 2] /= d;
}

// single-block exclusive scan of (deg+1) -> CSR offsets
__global__ void scan_deg(const float* __restrict__ deg, int* __restrict__ off, int N) {
  __shared__ int sh[1024];
  __shared__ int carry;
  int tid = threadIdx.x;
  if (tid == 0) carry = 0;
  __syncthreads();
  for (int base = 0; base < N; base += 1024) {
    int i = base + tid;
    int v = (i < N) ? ((int)deg[i] + 1) : 0;
    sh[tid] = v;
    __syncthreads();
    for (int o = 1; o < 1024; o <<= 1) {
      int t = (tid >= o) ? sh[tid - o] : 0;
      __syncthreads();
      sh[tid] += t;
      __syncthreads();
    }
    if (i < N) off[i] = carry + sh[tid] - v;
    __syncthreads();
    if (tid == 0) carry += sh[1023];
    __syncthreads();
  }
  if (threadIdx.x == 0) off[N] = carry;
}

__global__ void csr_self(const int* __restrict__ off, int* __restrict__ csr,
                         int* __restrict__ cursor, int N, int E) {
  int v = blockIdx.x * blockDim.x + threadIdx.x;
  if (v >= N) return;
  csr[off[v]] = E + v;     // self-loop edge id
  cursor[v] = 1;
}

__global__ void csr_fill(const int* __restrict__ dstA, const int* __restrict__ off,
                         int* __restrict__ cursor, int* __restrict__ csr, int E) {
  int e = blockIdx.x * blockDim.x + threadIdx.x;
  if (e >= E) return;
  int d = dstA[e];
  int p = atomicAdd(&cursor[d], 1);
  csr[off[d] + p] = e;
}

// ---------------- per-layer kernels ----------------
__global__ void dots_kernel(const float* __restrict__ hW, const float* __restrict__ as_,
                            const float* __restrict__ ad_, float* __restrict__ hs,
                            float* __restrict__ hd, int N, int C) {
  int wave = (blockIdx.x * blockDim.x + threadIdx.x) >> 5;
  int lane = threadIdx.x & 31;
  if (wave >= N) return;
  const float* row = hW + (size_t)wave * C;
  float s0 = 0.f, s1 = 0.f;
  for (int c = lane; c < C; c += 32) { float v = row[c]; s0 += v * as_[c]; s1 += v * ad_[c]; }
  for (int o = 16; o > 0; o >>= 1) { s0 += __shfl_down(s0, o); s1 += __shfl_down(s1, o); }
  if (lane == 0) { hs[wave] = s0; hd[wave] = s1; }
}

__global__ void w3_kernel(const float* __restrict__ We, const float* __restrict__ ae,
                          float* __restrict__ w3, int C) {
  int j = threadIdx.x;
  if (j < 3) {
    float s = 0.f;
    for (int c = 0; c < C; ++c) s += We[(size_t)j * C + c] * ae[c];
    w3[j] = s;
  }
}

__global__ void init_maxden(float* __restrict__ mx, float* __restrict__ den, int N) {
  int i = blockIdx.x * blockDim.x + threadIdx.x;
  if (i < N) { mx[i] = -INFINITY; den[i] = 0.0f; }
}

__global__ void edge_score(const int* __restrict__ srcA, const int* __restrict__ dstA,
                           const float* __restrict__ ea, const float* __restrict__ loops,
                           const float* __restrict__ hs, const float* __restrict__ hd,
                           const float* __restrict__ w3, float* __restrict__ s,
                           float* __restrict__ mx, int E, int N) {
  int e = blockIdx.x * blockDim.x + threadIdx.x;
  int Ea = E + N;
  if (e >= Ea) return;
  int sv, dv; float a0, a1, a2;
  if (e < E) {
    sv = srcA[e]; dv = dstA[e];
    a0 = ea[e * 3 + 0]; a1 = ea[e * 3 + 1]; a2 = ea[e * 3 + 2];
  } else {
    sv = dv = e - E;
    a0 = loops[dv * 3 + 0]; a1 = loops[dv * 3 + 1]; a2 = loops[dv * 3 + 2];
  }
  float v = hs[sv] + hd[dv] + a0 * w3[0] + a1 * w3[1] + a2 * w3[2];
  v = (v > 0.f) ? v : NEG_SLOPE * v;             // leaky relu
  s[e] = v;
  atomicMaxF(&mx[dv], v);
}

__global__ void soft_z(const int* __restrict__ dstA, float* __restrict__ s,
                       const float* __restrict__ mx, float* __restrict__ den, int E, int N) {
  int e = blockIdx.x * blockDim.x + threadIdx.x;
  int Ea = E + N;
  if (e >= Ea) return;
  int d = (e < E) ? dstA[e] : (e - E);
  float z = expf(s[e] - mx[d]);
  s[e] = z;                                      // z overwrites s
  atomicAdd(&den[d], z);
}

// CSR aggregation: out[node] = sum_e alpha_e * hW[src_e] + b ; optional relu; bf16 requant
__global__ void aggregate(const float* __restrict__ hW, const float* __restrict__ z,
                          const float* __restrict__ den, const int* __restrict__ off,
                          const int* __restrict__ csr, const int* __restrict__ srcA,
                          const float* __restrict__ bias, float* __restrict__ hout,
                          unsigned short* __restrict__ hbf, int N, int C, int E, int doRelu) {
  __shared__ int   s_src[256];
  __shared__ float s_w[256];
  int node = blockIdx.x;
  int tid = threadIdx.x;
  if (node >= N) {                               // pad rows -> zeros
    for (int c = tid; c < C; c += 256) {
      hout[(size_t)node * C + c] = 0.0f;
      hbf[(size_t)node * C + c] = f2bf(0.0f);
    }
    return;
  }
  int o0 = off[node], o1 = off[node + 1];
  float invden = 1.0f / (den[node] + 1e-16f);
  float acc[4] = {0.f, 0.f, 0.f, 0.f};
  for (int base = o0; base < o1; base += 256) {
    int cnt = min(256, o1 - base);
    if (tid < cnt) {
      int e = csr[base + tid];
      s_src[tid] = (e < E) ? srcA[e] : node;
      s_w[tid] = z[e] * invden;
    }
    __syncthreads();
    for (int j = 0; j < cnt; ++j) {
      const float* row = hW + (size_t)s_src[j] * C;
      float w = s_w[j];
#pragma unroll
      for (int r = 0; r < 4; ++r) { int c = tid + (r << 8); if (c < C) acc[r] += w * row[c]; }
    }
    __syncthreads();
  }
#pragma unroll
  for (int r = 0; r < 4; ++r) {
    int c = tid + (r << 8);
    if (c < C) {
      float v = acc[r] + bias[c];
      if (doRelu) v = fmaxf(v, 0.0f);
      hout[(size_t)node * C + c] = v;
      hbf[(size_t)node * C + c] = f2bf(v);
    }
  }
}

__global__ void pool_kernel(const float* __restrict__ h, const int* __restrict__ batch,
                            float* __restrict__ pooled, int N, int C) {
  int idx = blockIdx.x * blockDim.x + threadIdx.x;
  if (idx >= N * C) return;
  int node = idx / C, c = idx - node * C;
  atomicAdd(&pooled[(size_t)batch[node] * C + c], h[idx]);
}

// ---------------- host ----------------
extern "C" void kernel_launch(void* const* d_in, const int* in_sizes, int n_in,
                              void* d_out, int out_size, void* d_ws, size_t ws_size,
                              hipStream_t stream) {
  const float* x     = (const float*)d_in[0];
  const int*   ei    = (const int*)d_in[1];
  const float* eattr = (const float*)d_in[2];
  const int*   batch = (const int*)d_in[3];
  const int N  = in_sizes[0] / 3;
  const int E  = in_sizes[1] / 2;
  const int Bb = out_size / 1024;
  const int Ea = E + N;
  const int N16 = (N + 15) & ~15;
  const int* srcA = ei;
  const int* dstA = ei + E;
  const int dims[6] = {3, 64, 128, 256, 512, 1024};

  uintptr_t p = (uintptr_t)d_ws;
  auto alloc = [&](size_t bytes) -> void* {
    uintptr_t r = (p + 255) & ~(uintptr_t)255; p = r + bytes; return (void*)r;
  };
  unsigned short* hbf    = (unsigned short*)alloc((size_t)N16 * 1024 * 2);
  unsigned short* wtbf   = (unsigned short*)alloc((size_t)1024 * 1024 * 2);
  unsigned short* pbf    = (unsigned short*)alloc((size_t)Bb * 1024 * 2);
  float* hW     = (float*)alloc((size_t)N16 * 1024 * 4);
  float* hcur   = (float*)alloc((size_t)N16 * 1024 * 4);
  float* hs     = (float*)alloc((size_t)N * 4);
  float* hd     = (float*)alloc((size_t)N * 4);
  float* deg    = (float*)alloc((size_t)N * 4);
  float* loops  = (float*)alloc((size_t)N * 3 * 4);
  float* sbuf   = (float*)alloc((size_t)Ea * 4);
  float* mx     = (float*)alloc((size_t)N * 4);
  float* den    = (float*)alloc((size_t)N * 4);
  int*   off    = (int*)alloc((size_t)(N + 1) * 4);
  int*   cursor = (int*)alloc((size_t)N * 4);
  int*   csr    = (int*)alloc((size_t)Ea * 4);
  float* pooled = (float*)alloc((size_t)Bb * 1024 * 4);
  float* w3     = (float*)alloc(64);

  hipMemsetAsync(deg, 0, (size_t)N * 4, stream);
  hipMemsetAsync(loops, 0, (size_t)N * 3 * 4, stream);
  hipMemsetAsync(pooled, 0, (size_t)Bb * 1024 * 4, stream);

  const int T = 256;
  deg_loop<<<(E + T - 1) / T, T, 0, stream>>>(dstA, eattr, deg, loops, E);
  loop_div<<<(N + T - 1) / T, T, 0, stream>>>(loops, deg, N);
  scan_deg<<<1, 1024, 0, stream>>>(deg, off, N);
  csr_self<<<(N + T - 1) / T, T, 0, stream>>>(off, csr, cursor, N, E);
  csr_fill<<<(E + T - 1) / T, T, 0, stream>>>(dstA, off, cursor, csr, E);
  x_to_bf<<<(N16 * 32 + T - 1) / T, T, 0, stream>>>(x, hbf, N, N16);

  for (int i = 0; i < 5; ++i) {
    const int Cin = dims[i], Cout = dims[i + 1];
    const int Kp = (i == 0) ? 32 : Cin;
    const float* W   = (const float*)d_in[4 + 6 * i + 0];
    const float* as_ = (const float*)d_in[4 + 6 * i + 1];
    const float* ad_ = (const float*)d_in[4 + 6 * i + 2];
    const float* We  = (const float*)d_in[4 + 6 * i + 3];
    const float* ae  = (const float*)d_in[4 + 6 * i + 4];
    const float* b   = (const float*)d_in[4 + 6 * i + 5];

    wt_to_bf<<<(Cout * Kp + T - 1) / T, T, 0, stream>>>(W, wtbf, Cin, Kp, Cout);
    int tiles = (N16 / 16) * (Cout / 16);
    gemm_bf16_wmma<<<(tiles + 7) / 8, T, 0, stream>>>(hbf, wtbf, hW, N16, Kp, Cout, nullptr);
    dots_kernel<<<(N + 7) / 8, T, 0, stream>>>(hW, as_, ad_, hs, hd, N, Cout);
    w3_kernel<<<1, 4, 0, stream>>>(We, ae, w3, Cout);
    init_maxden<<<(N + T - 1) / T, T, 0, stream>>>(mx, den, N);
    edge_score<<<(Ea + T - 1) / T, T, 0, stream>>>(srcA, dstA, eattr, loops, hs, hd, w3,
                                                   sbuf, mx, E, N);
    soft_z<<<(Ea + T - 1) / T, T, 0, stream>>>(dstA, sbuf, mx, den, E, N);
    aggregate<<<N16, T, 0, stream>>>(hW, sbuf, den, off, csr, srcA, b, hcur, hbf,
                                     N, Cout, E, (i < 4) ? 1 : 0);
  }

  // global add pool + FC (WMMA, bias fused)
  pool_kernel<<<(N * 1024 + T - 1) / T, T, 0, stream>>>(hcur, batch, pooled, N, 1024);
  f_to_bf<<<(Bb * 1024 + T - 1) / T, T, 0, stream>>>(pooled, pbf, Bb * 1024);
  const float* fcw = (const float*)d_in[34];
  const float* fcb = (const float*)d_in[35];
  wt_to_bf<<<(1024 * 1024 + T - 1) / T, T, 0, stream>>>(fcw, wtbf, 1024, 1024, 1024);
  int ftiles = (Bb / 16) * (1024 / 16);
  gemm_bf16_wmma<<<(ftiles + 7) / 8, T, 0, stream>>>(pbf, wtbf, (float*)d_out,
                                                     Bb, 1024, 1024, fcb);
}
